// L2QAttention_61289183314182
// MI455X (gfx1250) — compile-verified, matrix operand order checked by reading, not run on hardware
//
#include <hip/hip_runtime.h>
#include <hip/hip_bf16.h>

// ---------------------------------------------------------------------------
// Quadratic-ReLU attention, fused flash-style, all GEMMs via
// v_wmma_f32_16x16x32_f16 (wave32 CDNA5 WMMA).
//
// Roofline: ~65 GFLOP total, ~110 MB HBM traffic as long as the [B,H,N,N]
// logits (402 MB) are never materialized -> attention is fused per 32-query
// tile with on-chip quadratic-ReLU + row-sum normalization.
//
// v2 changes vs v1:
//  - One-shot f32->f16 conversion pass for x / w_qkv / w_proj so the GEMM
//    K-loops load f16 directly (no v_cvt in hot loops, half the L2 bytes).
//  - Bigger per-wave tiles for operand reuse:
//      GEMMs:     32x64 output per wave  -> 8 WMMA per (2A + 4B) frag loads
//      attention: 32 queries per wave    -> K/V frags reused across 2 q-tiles
// ---------------------------------------------------------------------------

typedef __attribute__((ext_vector_type(16))) _Float16 v16h;
typedef __attribute__((ext_vector_type(8)))  float    v8f;

constexpr int DIMC   = 768;
constexpr int NHEADS = 12;
constexpr int HD     = 64;          // head dim
constexpr int BATCH  = 8;
constexpr int SEQ    = 1024;
constexpr int MROWS  = BATCH * SEQ; // 8192
constexpr int OCOLS  = 3 * DIMC;    // 2304
constexpr float SCALE = 0.125f;     // 64^-0.5

__device__ inline v8f zero8() {
    v8f z = {0.f, 0.f, 0.f, 0.f, 0.f, 0.f, 0.f, 0.f};
    return z;
}

// A/B fragment for V_WMMA_F32_16X16X32_F16 (16-bit operand layout):
// lane holds row (A: M, B: N) = lane%16; half = lane/16 selects which
// 8-element K chunks: K = kbase + {8*half .. +7} and {kbase+16+8*half .. +7}.
__device__ inline v16h load_frag_f16(const _Float16* __restrict__ p, int ld,
                                     int row, int kbase, int half) {
    v16h f;
    const _Float16* r = p + (size_t)row * ld + kbase + 8 * half;
#pragma unroll
    for (int e = 0; e < 8; ++e) f[e] = r[e];
#pragma unroll
    for (int e = 0; e < 8; ++e) f[8 + e] = r[16 + e];
    return f;
}

// ---------------------------------------------------------------------------
// Kernel 0: one-shot f32 -> f16 conversion (8 elems/thread, vectorizable).
// ---------------------------------------------------------------------------
__global__ void cvt_f16_kernel(const float* __restrict__ src,
                               _Float16* __restrict__ dst, int n8) {
    const int i = blockIdx.x * blockDim.x + threadIdx.x;
    if (i >= n8) return;
    const size_t base = (size_t)i * 8;
#pragma unroll
    for (int e = 0; e < 8; ++e) dst[base + e] = (_Float16)src[base + e];
}

// ---------------------------------------------------------------------------
// Kernel 1: qkv = x @ w_qkv^T ; scatter into Q [B,H,N,d], K [B,H,N,d],
// Vt [B,H,d,N] (transposed so attn@v B-operand is K-contiguous), all f16.
// One wave per 32x64 output tile: per K-step, 2 A-frags + 4 B-frags -> 8 WMMA.
// Each 64-wide output span maps to exactly one (s, h): s/h uniform per wave.
// ---------------------------------------------------------------------------
__global__ void qkv_gemm_kernel(const _Float16* __restrict__ xh,
                                const _Float16* __restrict__ wh,
                                _Float16* __restrict__ qb,
                                _Float16* __restrict__ kb,
                                _Float16* __restrict__ vtb) {
    const int wid  = threadIdx.x >> 5;
    const int lane = threadIdx.x & 31;
    const int tile = blockIdx.x * (blockDim.x >> 5) + wid;
    const int mt_count = MROWS / 32;   // 256
    const int ot_count = OCOLS / 64;   // 36
    if (tile >= mt_count * ot_count) return;
    const int mt = tile / ot_count;
    const int ot = tile % ot_count;
    const int half = lane >> 4;
    const int col  = lane & 15;

    v8f acc[2][4];
#pragma unroll
    for (int mi = 0; mi < 2; ++mi)
#pragma unroll
        for (int ni = 0; ni < 4; ++ni) acc[mi][ni] = zero8();

    for (int k = 0; k < DIMC; k += 32) {
        v16h a0 = load_frag_f16(xh, DIMC, mt * 32 + col,      k, half);
        v16h a1 = load_frag_f16(xh, DIMC, mt * 32 + 16 + col, k, half);
#pragma unroll
        for (int ni = 0; ni < 4; ++ni) {
            v16h b = load_frag_f16(wh, DIMC, ot * 64 + ni * 16 + col, k, half);
            acc[0][ni] = __builtin_amdgcn_wmma_f32_16x16x32_f16(
                false, a0, false, b, (short)0, acc[0][ni], false, false);
            acc[1][ni] = __builtin_amdgcn_wmma_f32_16x16x32_f16(
                false, a1, false, b, (short)0, acc[1][ni], false, false);
        }
    }

    // C layout: lane holds D[M = r + 8*half][N = col] per 16x16 sub-tile.
    const int s = (ot * 64) / DIMC;   // 0=q, 1=k, 2=v   (uniform per wave)
    const int h = ot % 12;            // head             (uniform per wave)
#pragma unroll
    for (int mi = 0; mi < 2; ++mi) {
#pragma unroll
        for (int ni = 0; ni < 4; ++ni) {
            const int dd = ni * 16 + col;
#pragma unroll
            for (int r = 0; r < 8; ++r) {
                const int m  = mt * 32 + mi * 16 + r + 8 * half;
                const int bi = m >> 10;
                const int n  = m & 1023;
                const size_t bh = (size_t)(bi * NHEADS + h);
                const _Float16 v = (_Float16)acc[mi][ni][r];
                if (s == 0)      qb[(bh * SEQ + n) * HD + dd] = v;
                else if (s == 1) kb[(bh * SEQ + n) * HD + dd] = v;
                else             vtb[(bh * HD + dd) * SEQ + n] = v;
            }
        }
    }
}

// ---------------------------------------------------------------------------
// Kernel 2: fused quadratic-ReLU attention, 32 queries per wave.
// Per 32-key chunk: 8 WMMA for logits (K/V frags shared across both q-tiles),
// quadratic-ReLU on f32 accumulators, P round-trips through LDS (same-wave DS
// is in-order -> no barrier), 8 WMMA for P@V. Normalize by row-sum at end.
// ---------------------------------------------------------------------------
__global__ void attn_kernel(const _Float16* __restrict__ qb,
                            const _Float16* __restrict__ kb,
                            const _Float16* __restrict__ vtb,
                            const float* __restrict__ alpha,
                            const float* __restrict__ beta,
                            const float* __restrict__ gamma,
                            _Float16* __restrict__ aout) {
    __shared__ _Float16 lds[8][32][40];   // [wave][M (32 queries)][K] padded

    const int wid  = threadIdx.x >> 5;
    const int lane = threadIdx.x & 31;
    const int half = lane >> 4;
    const int col  = lane & 15;

    const int tiles_per_bh = SEQ / 32;    // 32
    const int w = blockIdx.x * (blockDim.x >> 5) + wid;
    if (w >= BATCH * NHEADS * tiles_per_bh) return;
    const int bh = w / tiles_per_bh;      // b*NHEADS + h
    const int nt = w % tiles_per_bh;
    const int h  = bh % NHEADS;
    const int b  = bh / NHEADS;

    const float a2 = alpha[h] * SCALE * SCALE;  // fold SCALE into polynomial
    const float b1 = beta[h] * SCALE;
    const float g0 = gamma[h];

    const _Float16* qbase = qb  + (size_t)bh * SEQ * HD;
    const _Float16* kbase = kb  + (size_t)bh * SEQ * HD;
    const _Float16* vbase = vtb + (size_t)bh * HD * SEQ;

    // q A-fragments for both 16-row tiles, loaded once (K = d = 64).
    v16h qA[2][2];
#pragma unroll
    for (int mi = 0; mi < 2; ++mi) {
        const int qrow = nt * 32 + mi * 16 + col;
        qA[mi][0] = load_frag_f16(qbase, HD, qrow, 0, half);
        qA[mi][1] = load_frag_f16(qbase, HD, qrow, 32, half);
    }

    v8f oacc[2][4];
#pragma unroll
    for (int mi = 0; mi < 2; ++mi)
#pragma unroll
        for (int d = 0; d < 4; ++d) oacc[mi][d] = zero8();
    v8f den[2] = {zero8(), zero8()};

    for (int k0 = 0; k0 < SEQ; k0 += 32) {
#pragma unroll
        for (int t = 0; t < 2; ++t) {
            const int krow = k0 + t * 16 + col;     // key row (B operand N)
            v16h kB0 = load_frag_f16(kbase, HD, krow, 0, half);
            v16h kB1 = load_frag_f16(kbase, HD, krow, 32, half);
#pragma unroll
            for (int mi = 0; mi < 2; ++mi) {
                v8f s = zero8();
                s = __builtin_amdgcn_wmma_f32_16x16x32_f16(
                    false, qA[mi][0], false, kB0, (short)0, s, false, false);
                s = __builtin_amdgcn_wmma_f32_16x16x32_f16(
                    false, qA[mi][1], false, kB1, (short)0, s, false, false);
#pragma unroll
                for (int r = 0; r < 8; ++r) {
                    const float sv = s[r];
                    float p = a2 * sv * sv + b1 * sv + g0;
                    p = p > 0.f ? p : 0.f;          // v_max_num_f32
                    den[mi][r] += p;
                    lds[wid][mi * 16 + r + 8 * half][t * 16 + col] = (_Float16)p;
                }
            }
        }
        // V B-fragments for this key chunk (shared by both q-tiles).
        v16h vB[4];
#pragma unroll
        for (int d = 0; d < 4; ++d)
            vB[d] = load_frag_f16(vbase, SEQ, d * 16 + col, k0, half);
        // Re-read P as A-fragments (row = mi*16 + lane%16, K = 32 keys).
        // Same-wave DS ops are in-order -> no barrier needed.
#pragma unroll
        for (int mi = 0; mi < 2; ++mi) {
            v16h pA;
            const _Float16* rp = &lds[wid][mi * 16 + col][8 * half];
#pragma unroll
            for (int e = 0; e < 8; ++e) pA[e] = rp[e];
#pragma unroll
            for (int e = 0; e < 8; ++e) pA[8 + e] = rp[16 + e];
#pragma unroll
            for (int d = 0; d < 4; ++d)
                oacc[mi][d] = __builtin_amdgcn_wmma_f32_16x16x32_f16(
                    false, pA, false, vB[d], (short)0, oacc[mi][d], false, false);
        }
    }

    // Row denominators: lane holds rows M=r+8*half with partial sums over
    // cols == lane%16 -> butterfly-reduce across the 16 lanes of each half.
#pragma unroll
    for (int mi = 0; mi < 2; ++mi) {
#pragma unroll
        for (int r = 0; r < 8; ++r) {
            float d = den[mi][r];
            d += __shfl_xor(d, 1, 16);
            d += __shfl_xor(d, 2, 16);
            d += __shfl_xor(d, 4, 16);
            d += __shfl_xor(d, 8, 16);
            den[mi][r] = 1.0f / (d + 1e-6f);
        }
    }

    // Normalize and store f16 in [B*N, C] layout for the proj GEMM.
#pragma unroll
    for (int mi = 0; mi < 2; ++mi) {
#pragma unroll
        for (int r = 0; r < 8; ++r) {
            const int n = nt * 32 + mi * 16 + r + 8 * half;
            _Float16* dst = aout + ((size_t)(b * SEQ + n)) * DIMC + h * HD + col;
            const float inv = den[mi][r];
#pragma unroll
            for (int d = 0; d < 4; ++d)
                dst[d * 16] = (_Float16)(oacc[mi][d][r] * inv);
        }
    }
}

// ---------------------------------------------------------------------------
// Kernel 3: out = attn_out @ w_proj^T + b_proj  (fp32 output), 32x64 per wave.
// ---------------------------------------------------------------------------
__global__ void proj_gemm_kernel(const _Float16* __restrict__ aout,
                                 const _Float16* __restrict__ wph,
                                 const float* __restrict__ bias,
                                 float* __restrict__ out) {
    const int wid  = threadIdx.x >> 5;
    const int lane = threadIdx.x & 31;
    const int tile = blockIdx.x * (blockDim.x >> 5) + wid;
    const int mt_count = MROWS / 32;   // 256
    const int ot_count = DIMC / 64;    // 12
    if (tile >= mt_count * ot_count) return;
    const int mt = tile / ot_count;
    const int ot = tile % ot_count;
    const int half = lane >> 4;
    const int col  = lane & 15;

    v8f acc[2][4];
#pragma unroll
    for (int mi = 0; mi < 2; ++mi)
#pragma unroll
        for (int ni = 0; ni < 4; ++ni) acc[mi][ni] = zero8();

    for (int k = 0; k < DIMC; k += 32) {
        v16h a0 = load_frag_f16(aout, DIMC, mt * 32 + col,      k, half);
        v16h a1 = load_frag_f16(aout, DIMC, mt * 32 + 16 + col, k, half);
#pragma unroll
        for (int ni = 0; ni < 4; ++ni) {
            v16h bfr = load_frag_f16(wph, DIMC, ot * 64 + ni * 16 + col, k, half);
            acc[0][ni] = __builtin_amdgcn_wmma_f32_16x16x32_f16(
                false, a0, false, bfr, (short)0, acc[0][ni], false, false);
            acc[1][ni] = __builtin_amdgcn_wmma_f32_16x16x32_f16(
                false, a1, false, bfr, (short)0, acc[1][ni], false, false);
        }
    }

#pragma unroll
    for (int mi = 0; mi < 2; ++mi) {
#pragma unroll
        for (int ni = 0; ni < 4; ++ni) {
            const int o  = ot * 64 + ni * 16 + col;
            const float bv = bias[o];
#pragma unroll
            for (int r = 0; r < 8; ++r) {
                const int m = mt * 32 + mi * 16 + r + 8 * half;
                out[(size_t)m * DIMC + o] = acc[mi][ni][r] + bv;
            }
        }
    }
}

// ---------------------------------------------------------------------------
extern "C" void kernel_launch(void* const* d_in, const int* in_sizes, int n_in,
                              void* d_out, int out_size, void* d_ws, size_t ws_size,
                              hipStream_t stream) {
    const float* x      = (const float*)d_in[0];
    const float* w_qkv  = (const float*)d_in[1];
    const float* w_proj = (const float*)d_in[2];
    const float* b_proj = (const float*)d_in[3];
    const float* alpha  = (const float*)d_in[4];
    const float* beta   = (const float*)d_in[5];
    const float* gamma  = (const float*)d_in[6];
    float* out = (float*)d_out;

    const size_t seg    = (size_t)MROWS * DIMC;   // 6,291,456
    const size_t n_x    = seg;                    // x elems
    const size_t n_wqkv = (size_t)OCOLS * DIMC;   // 1,769,472
    const size_t n_wprj = (size_t)DIMC * DIMC;    //   589,824

    _Float16* xh    = (_Float16*)d_ws;
    _Float16* wqkvh = xh + n_x;
    _Float16* wprjh = wqkvh + n_wqkv;
    _Float16* qb    = wprjh + n_wprj;
    _Float16* kb    = qb + seg;
    _Float16* vtb   = kb + seg;
    _Float16* aout  = vtb + seg;                  // total ~67.6 MB workspace

    // One-shot f32 -> f16 conversions (8 elems per thread).
    cvt_f16_kernel<<<(int)(n_x / 8 / 256), 256, 0, stream>>>(x, xh, (int)(n_x / 8));
    cvt_f16_kernel<<<(int)(n_wqkv / 8 + 255) / 256, 256, 0, stream>>>(w_qkv, wqkvh, (int)(n_wqkv / 8));
    cvt_f16_kernel<<<(int)(n_wprj / 8 + 255) / 256, 256, 0, stream>>>(w_proj, wprjh, (int)(n_wprj / 8));

    // 32x64 tile per wave32, 8 waves (256 threads) per block.
    const int qkv_tiles  = (MROWS / 32) * (OCOLS / 64);          // 9216
    const int attn_waves = BATCH * NHEADS * (SEQ / 32);          // 3072
    const int proj_tiles = (MROWS / 32) * (DIMC / 64);           // 3072

    qkv_gemm_kernel<<<(qkv_tiles + 7) / 8, 256, 0, stream>>>(xh, wqkvh, qb, kb, vtb);
    attn_kernel<<<(attn_waves + 7) / 8, 256, 0, stream>>>(qb, kb, vtb,
                                                          alpha, beta, gamma, aout);
    proj_gemm_kernel<<<(proj_tiles + 7) / 8, 256, 0, stream>>>(aout, wprjh, b_proj, out);
}
